// ODE_RNN_51436528337547
// MI455X (gfx1250) — compile-verified, matrix operand order checked by reading, not run on hardware
//
#include <hip/hip_runtime.h>
#include <hip/hip_bf16.h>

#define NWG      128
#define NTHREADS 256
#define NWAVES   8
#define HSZ      2048
#define IDIM     64
#define ODIM     64
#define NSTEPS   256
#define RKK      8

typedef __bf16 bf16;
typedef __attribute__((ext_vector_type(16))) __bf16 v16bf;
typedef __attribute__((ext_vector_type(8)))  float  v8f;
typedef __attribute__((ext_vector_type(4)))  unsigned u32x4;

union ABFrag { v16bf v; unsigned u[8]; u32x4 q[2]; };

// fp32 epilogue modes
#define OF_NONE  0
#define OF_PLAIN 1
#define OF_TANH  2
// bf16 (next-stage B vector) epilogue modes
#define OB_NONE  0
#define OB_PLAIN 1
#define OB_TANH  2
#define OB_AXPY  3   // (bf16)(haux[row] + hc * sum)

// ---------------- grid-wide sense-reversing barrier (persistent kernel) ----
__device__ __forceinline__ void grid_sync(unsigned* cnt, unsigned* gen) {
  __syncthreads();
  if (threadIdx.x == 0) {
    __threadfence();
    unsigned g = __atomic_load_n(gen, __ATOMIC_RELAXED);
    unsigned prev = __atomic_fetch_add(cnt, 1u, __ATOMIC_ACQ_REL);
    if (prev == NWG - 1u) {
      __atomic_store_n(cnt, 0u, __ATOMIC_RELAXED);
      __atomic_fetch_add(gen, 1u, __ATOMIC_RELEASE);
    } else {
      while (__atomic_load_n(gen, __ATOMIC_ACQUIRE) == g) {
        __builtin_amdgcn_s_sleep(1);
      }
    }
    __threadfence();
  }
  __syncthreads();
}

// --------- index of element (r,k) inside the WMMA-A-tiled bf16 weight ------
// Tile = 16 rows x 32 cols = 512 elems (1KB); lane L owns 16 contiguous elems.
// ISA 16-bit A 16x32 layout: lanes 0-15 rows, lanes 16-31 same rows / K+8;
// VGPR v holds K pair base = (v<4 ? 2v : 16+2(v-4)) + 8*half.
__device__ __forceinline__ size_t tile_index(int r, int k, int C) {
  int s    = r >> 4, rloc = r & 15;
  int c    = k >> 5, koff = k & 31;
  int half = (koff >> 3) & 1;
  int vv   = ((koff & 16) >> 2) + ((koff & 7) >> 1);
  int lane = rloc + (half << 4);
  return ((size_t)s * (size_t)(C >> 5) + (size_t)c) * 512u
         + (size_t)lane * 16u + (size_t)vv * 2u + (size_t)(koff & 1);
}

__device__ void conv_tiled(const float* __restrict__ src, bf16* __restrict__ dst,
                           int R, int C, int gtid, int gsz) {
  size_t n = (size_t)R * (size_t)C;
  for (size_t i = gtid; i < n; i += (size_t)gsz) {
    int r = (int)(i / (size_t)C), k = (int)(i % (size_t)C);
    dst[tile_index(r, k, C)] = (bf16)src[i];
  }
}

// ---------------- 16-row-strip GEMV via v_wmma_f32_16x16x32_bf16 ------------
__device__ void gemv_wmma(const bf16* __restrict__ Wt, int R, int C,
                          const bf16* __restrict__ vb,
                          float* outf, int ofmode,
                          bf16* outb, int obmode,
                          const float* haux, float hc,
                          float* red) {
  const int  lane  = threadIdx.x & 31;
  const int  wave  = threadIdx.x >> 5;
  const int  half  = lane >> 4;
  const int  nstr  = R >> 4;
  const int  nchk  = C >> 5;
  const bool col0  = ((lane & 15) == 0);       // only these lanes carry B data
  const bf16* vhb  = vb + half * 8;

  for (int s = blockIdx.x; s < nstr; s += NWG) {
    const bf16* tbase = Wt + (size_t)s * (size_t)nchk * 512u + (size_t)lane * 16u;
    v8f acc = {};
    ABFrag b0, b1;                              // zero once; dead lanes stay zero
    b0.q[0] = (u32x4){0u, 0u, 0u, 0u}; b0.q[1] = (u32x4){0u, 0u, 0u, 0u};
    b1.q[0] = (u32x4){0u, 0u, 0u, 0u}; b1.q[1] = (u32x4){0u, 0u, 0u, 0u};

    int c = wave;
    // two chunks per iteration: all 4 loads issued before the 2 WMMAs so the
    // second chunk's loads stay in flight under the first WMMA (ILP, 1 wave/SIMD)
    for (; c + NWAVES < nchk; c += 2 * NWAVES) {
      const bf16* tp0 = tbase + (size_t)c * 512u;
      const bf16* tp1 = tp0 + (size_t)NWAVES * 512u;
      __builtin_prefetch(tp1 + NWAVES * 512, 0, 1);   // speculative, unguarded
      ABFrag x0, x1;
      x0.q[0] = *(const u32x4*)(tp0);                 // global_load_b128 x4
      x0.q[1] = *(const u32x4*)(tp0 + 8);
      x1.q[0] = *(const u32x4*)(tp1);
      x1.q[1] = *(const u32x4*)(tp1 + 8);
      if (col0) {                                     // exec-masked: 2 lanes
        const bf16* vp0 = vhb + c * 32;
        const bf16* vp1 = vp0 + NWAVES * 32;
        b0.q[0] = *(const u32x4*)(vp0); b0.q[1] = *(const u32x4*)(vp0 + 16);
        b1.q[0] = *(const u32x4*)(vp1); b1.q[1] = *(const u32x4*)(vp1 + 16);
      }
      // reconverged -> EXEC all-1s for matrix ops
      acc = __builtin_amdgcn_wmma_f32_16x16x32_bf16(
          false, x0.v, false, b0.v, (short)0, acc, false, false);
      acc = __builtin_amdgcn_wmma_f32_16x16x32_bf16(
          false, x1.v, false, b1.v, (short)0, acc, false, false);
    }
    if (c < nchk) {                                   // remainder chunk
      const bf16* tp0 = tbase + (size_t)c * 512u;
      ABFrag x0;
      x0.q[0] = *(const u32x4*)(tp0);
      x0.q[1] = *(const u32x4*)(tp0 + 8);
      if (col0) {
        const bf16* vp0 = vhb + c * 32;
        b0.q[0] = *(const u32x4*)(vp0); b0.q[1] = *(const u32x4*)(vp0 + 16);
      }
      acc = __builtin_amdgcn_wmma_f32_16x16x32_bf16(
          false, x0.v, false, b0.v, (short)0, acc, false, false);
    }

    // column 0 of C tile: lane 0 holds M=0..7, lane 16 holds M=8..15
    if (col0) {
      float* dst = red + wave * 16 + half * 8;
#pragma unroll
      for (int r = 0; r < 8; ++r) dst[r] = acc[r];
    }
    __syncthreads();
    if (threadIdx.x < 16) {
      float sum = 0.0f;
#pragma unroll
      for (int j = 0; j < NWAVES; ++j) sum += red[j * 16 + threadIdx.x];
      const int row = (s << 4) + threadIdx.x;
      if (ofmode == OF_PLAIN)      outf[row] = sum;
      else if (ofmode == OF_TANH)  outf[row] = tanhf(sum);
      if (obmode == OB_PLAIN)      outb[row] = (bf16)sum;
      else if (obmode == OB_TANH)  outb[row] = (bf16)tanhf(sum);
      else if (obmode == OB_AXPY)  outb[row] = (bf16)fmaf(hc, sum, haux[row]);
    }
    __syncthreads();
  }
}

// ---------------- persistent ODE-RNN kernel --------------------------------
__global__ void __launch_bounds__(NTHREADS)
odernn_persistent(const float* __restrict__ t_arr, const float* __restrict__ x,
                  const float* __restrict__ W1f, const float* __restrict__ W2f,
                  const float* __restrict__ Whf, const float* __restrict__ Wdf,
                  const float* __restrict__ Wihf, const float* __restrict__ Whhf,
                  float* __restrict__ out, unsigned char* __restrict__ ws) {
  __shared__ float red[NWAVES * 16];

  unsigned* bar_cnt = (unsigned*)ws;
  unsigned* bar_gen = (unsigned*)(ws + 64);

  // fp32 state: h, k1..k4, gi(3H), gh(3H)  -> 11*H floats
  float* F  = (float*)(ws + 256);
  float* h  = F;
  float* k1 = F + 1 * HSZ;
  float* k2 = F + 2 * HSZ;
  float* k3 = F + 3 * HSZ;
  float* k4 = F + 4 * HSZ;
  float* gi = F + 5 * HSZ;                   // 3H
  float* gh = F + 8 * HSZ;                   // 3H
  // bf16 prepacked B vectors (start 256-aligned: 256 + 11*2048*4 = 90368)
  bf16* V    = (bf16*)(ws + 90368);
  bf16* hbf  = V;                            // (bf16)h
  bf16* ubf  = hbf + HSZ;                    // (bf16)tanh(W1@...)
  bf16* hkbf = ubf + HSZ;                    // (bf16)(h + c*k_i)
  bf16* hdbf = hkbf + HSZ;                   // (bf16)(W_hidden@h)
  bf16* xbf  = hdbf + HSZ;                   // all inputs, N*I
  // tiled bf16 weights (start 139520, 256-aligned)
  bf16* W1t  = (bf16*)(ws + 139520);
  bf16* W2t  = W1t  + (size_t)HSZ * HSZ;
  bf16* Wht  = W2t  + (size_t)HSZ * HSZ;
  bf16* Whht = Wht  + (size_t)HSZ * HSZ;     // 3H x H
  bf16* Wdt  = Whht + (size_t)3 * HSZ * HSZ; // O x H
  bf16* Wiht = Wdt  + (size_t)ODIM * HSZ;    // 3H x I

  const int gtid = blockIdx.x * NTHREADS + threadIdx.x;
  const int gsz  = NWG * NTHREADS;

  // ---- one-time prep: tiled fp32->bf16 weights; bf16 inputs; h = 0
  conv_tiled(W1f,  W1t,  HSZ,     HSZ,  gtid, gsz);
  conv_tiled(W2f,  W2t,  HSZ,     HSZ,  gtid, gsz);
  conv_tiled(Whf,  Wht,  HSZ,     HSZ,  gtid, gsz);
  conv_tiled(Whhf, Whht, 3 * HSZ, HSZ,  gtid, gsz);
  conv_tiled(Wdf,  Wdt,  ODIM,    HSZ,  gtid, gsz);
  conv_tiled(Wihf, Wiht, 3 * HSZ, IDIM, gtid, gsz);
  for (int i = gtid; i < NSTEPS * IDIM; i += gsz) xbf[i] = (bf16)x[i];
  for (int i = gtid; i < HSZ; i += gsz) { h[i] = 0.0f; hbf[i] = (bf16)0.0f; }
  grid_sync(bar_cnt, bar_gen);

  float tprev = 0.0f;
  for (int n = 0; n < NSTEPS; ++n) {
    const float t1 = t_arr[n];
    const float dt = (t1 - tprev) * (1.0f / (float)RKK);
    tprev = t1;

    // ---- RK4 x 8 substeps. Producers prepack next stage's bf16 B vector.
    for (int s = 0; s < RKK; ++s) {
      gemv_wmma(W1t, HSZ, HSZ, hbf,  nullptr, OF_NONE,  ubf,  OB_TANH, h, 0.0f,      red); grid_sync(bar_cnt, bar_gen);
      gemv_wmma(W2t, HSZ, HSZ, ubf,  k1, OF_PLAIN, hkbf, OB_AXPY, h, 0.5f * dt, red); grid_sync(bar_cnt, bar_gen);
      gemv_wmma(W1t, HSZ, HSZ, hkbf, nullptr, OF_NONE,  ubf,  OB_TANH, h, 0.0f,      red); grid_sync(bar_cnt, bar_gen);
      gemv_wmma(W2t, HSZ, HSZ, ubf,  k2, OF_PLAIN, hkbf, OB_AXPY, h, 0.5f * dt, red); grid_sync(bar_cnt, bar_gen);
      gemv_wmma(W1t, HSZ, HSZ, hkbf, nullptr, OF_NONE,  ubf,  OB_TANH, h, 0.0f,      red); grid_sync(bar_cnt, bar_gen);
      gemv_wmma(W2t, HSZ, HSZ, ubf,  k3, OF_PLAIN, hkbf, OB_AXPY, h, dt,        red); grid_sync(bar_cnt, bar_gen);
      gemv_wmma(W1t, HSZ, HSZ, hkbf, nullptr, OF_NONE,  ubf,  OB_TANH, h, 0.0f,      red); grid_sync(bar_cnt, bar_gen);
      gemv_wmma(W2t, HSZ, HSZ, ubf,  k4, OF_PLAIN, nullptr, OB_NONE, h, 0.0f,   red); grid_sync(bar_cnt, bar_gen);
      for (int i = gtid; i < HSZ; i += gsz) {
        float hn = fmaf(dt * (1.0f / 6.0f),
                        k1[i] + 2.0f * k2[i] + 2.0f * k3[i] + k4[i], h[i]);
        h[i] = hn; hbf[i] = (bf16)hn;
      }
      grid_sync(bar_cnt, bar_gen);
    }

    // ---- decode: out[n] = tanh(W_dec @ (W_hidden @ h))
    gemv_wmma(Wht, HSZ, HSZ, hbf, nullptr, OF_NONE, hdbf, OB_PLAIN, h, 0.0f, red); grid_sync(bar_cnt, bar_gen);
    gemv_wmma(Wdt, ODIM, HSZ, hdbf, out + (size_t)n * ODIM, OF_TANH,
              nullptr, OB_NONE, h, 0.0f, red);                                     grid_sync(bar_cnt, bar_gen);

    // ---- GRU (h0==0 makes the step-0 zero-hidden special case a no-op)
    gemv_wmma(Wiht, 3 * HSZ, IDIM, xbf + n * IDIM, gi, OF_PLAIN,
              nullptr, OB_NONE, h, 0.0f, red);                                     grid_sync(bar_cnt, bar_gen);
    gemv_wmma(Whht, 3 * HSZ, HSZ, hbf, gh, OF_PLAIN,
              nullptr, OB_NONE, h, 0.0f, red);                                     grid_sync(bar_cnt, bar_gen);
    for (int i = gtid; i < HSZ; i += gsz) {
      float r  = 1.0f / (1.0f + expf(-(gi[i]           + gh[i])));
      float z  = 1.0f / (1.0f + expf(-(gi[HSZ + i]     + gh[HSZ + i])));
      float nn = tanhf(gi[2 * HSZ + i] + r * gh[2 * HSZ + i]);
      float hnew = (1.0f - z) * nn + z * h[i];
      h[i] = hnew; hbf[i] = (bf16)hnew;
    }
    grid_sync(bar_cnt, bar_gen);
  }
}

extern "C" void kernel_launch(void* const* d_in, const int* in_sizes, int n_in,
                              void* d_out, int out_size, void* d_ws, size_t ws_size,
                              hipStream_t stream) {
  (void)in_sizes; (void)n_in; (void)out_size; (void)ws_size;
  hipMemsetAsync(d_ws, 0, 256, stream);   // zero grid-barrier state (capturable)
  odernn_persistent<<<NWG, NTHREADS, 0, stream>>>(
      (const float*)d_in[0],  // t
      (const float*)d_in[1],  // x
      (const float*)d_in[2],  // W_ode1
      (const float*)d_in[3],  // W_ode2
      (const float*)d_in[4],  // W_hidden
      (const float*)d_in[5],  // W_dec
      (const float*)d_in[6],  // W_ih
      (const float*)d_in[7],  // W_hh
      (float*)d_out, (unsigned char*)d_ws);
}
